// EnhancedGraphSAGE_77747497992437
// MI455X (gfx1250) — matrix operand order repaired and checked by reading, not compiled, not run on hardware
//
#include <hip/hip_runtime.h>
#include <hip/hip_bf16.h>

typedef __attribute__((ext_vector_type(16))) __bf16 v16bf;
typedef __attribute__((ext_vector_type(8)))  float  v8f;

#define N_NODES 100000
#define N_EDGES 800000
#define BGRAPH  64
#define HDIM    128
#define LAYERS  3
#define NCLS    8
#define HTDIM   64
#define PDIM    320

struct Q2 { uint4 x, y; };   // 32 bytes, bit-castable to v16bf

__device__ __forceinline__ unsigned short f2bf_us(float f) {
    unsigned u = __float_as_uint(f);
    unsigned r = (u + 0x7FFFu + ((u >> 16) & 1u)) >> 16;   // round-to-nearest-even
    return (unsigned short)r;
}
// monotonic float<->uint transform for atomicMax on floats
__device__ __forceinline__ unsigned f2o(float f) {
    unsigned u = __float_as_uint(f);
    return (u & 0x80000000u) ? ~u : (u | 0x80000000u);
}
__device__ __forceinline__ float o2f(unsigned t) {
    unsigned bits = (t & 0x80000000u) ? (t & 0x7FFFFFFFu) : ~t;
    return __uint_as_float(bits);
}

// ---------------- node encoder: h = relu(x @ enc_W + enc_b) ----------------
__global__ void k_encoder(const float* __restrict__ x, const float* __restrict__ W,
                          const float* __restrict__ b, float* __restrict__ h) {
    int gid = blockIdx.x * blockDim.x + threadIdx.x;
    if (gid >= N_NODES * HDIM) return;
    int n = gid >> 7, j = gid & 127;
    const float* xr = x + n * 4;
    float acc = b[j] + xr[0] * W[0 * HDIM + j] + xr[1] * W[1 * HDIM + j]
                     + xr[2] * W[2 * HDIM + j] + xr[3] * W[3 * HDIM + j];
    h[gid] = fmaxf(acc, 0.f);
}

// ---------------- in-degree ----------------
__global__ void k_deg(const int* __restrict__ dst, float* __restrict__ deg) {
    int e = blockIdx.x * blockDim.x + threadIdx.x;
    if (e < N_EDGES) atomicAdd(&deg[dst[e]], 1.f);
}

// ------- build bf16 [Wl; Wr] TRANSPOSED: Wcat_t[col][k], col<128, k<256 -------
__global__ void k_wcat(const float* __restrict__ Wl, const float* __restrict__ Wr,
                       unsigned short* __restrict__ Wcat_t) {
    int gid = blockIdx.x * blockDim.x + threadIdx.x;   // 256*128
    int k = gid >> 7, j = gid & 127;
    float v = (k < HDIM) ? Wl[k * HDIM + j] : Wr[(k - HDIM) * HDIM + j];
    Wcat_t[j * 256 + k] = f2bf_us(v);
}

// ---------------- edge scatter: agg[dst] += h[src], 8 floats per thread ----------------
__global__ void k_scatter(const int* __restrict__ src, const int* __restrict__ dst,
                          const float* __restrict__ h, float* __restrict__ agg) {
    int gid = blockIdx.x * blockDim.x + threadIdx.x;   // E * 16 chunks
    if (gid >= N_EDGES * 16) return;
    int e = gid >> 4;
    int c = (gid & 15) * 8;
    int s = src[e], d = dst[e];
    const float4 v0 = *(const float4*)(h + (size_t)s * HDIM + c);
    const float4 v1 = *(const float4*)(h + (size_t)s * HDIM + c + 4);
    float* ap = agg + (size_t)d * HDIM + c;
    atomicAdd(ap + 0, v0.x); atomicAdd(ap + 1, v0.y);
    atomicAdd(ap + 2, v0.z); atomicAdd(ap + 3, v0.w);
    atomicAdd(ap + 4, v1.x); atomicAdd(ap + 5, v1.y);
    atomicAdd(ap + 6, v1.z); atomicAdd(ap + 7, v1.w);
}

// ------- fused SAGE layer: hn = [agg/deg | h] @ Wcat + bl; h_out = h + relu(LN(hn)) -------
// block = 256 threads (8 waves). Block computes 16 rows x 128 cols.
// Wave w owns column tile [16w, 16w+16). K = 256 done in 8 WMMA steps of 32.
// A fragment = 2x ds_load_b128 per step; B fragment = 2x global_load_b128 per step.
__global__ __launch_bounds__(256) void k_sage(
    const float* __restrict__ h_in, const float* __restrict__ agg,
    const float* __restrict__ deg, const unsigned short* __restrict__ Wcat_t,
    const float* __restrict__ bl, const float* __restrict__ lng,
    const float* __restrict__ lnb, float* __restrict__ h_out) {
    __shared__ unsigned short Abf[16][256];   // A tile, bf16 bits (row-major, K contiguous)
    __shared__ float Cls[16][HDIM];           // pre-LN result
    __shared__ float mu_s[16], rs_s[16];

    const int t = threadIdx.x;
    const int r0 = blockIdx.x * 16;           // N is a multiple of 16

    // stage A tile: cols [0,128) = agg/deg, cols [128,256) = h
    for (int q = 0; q < 16; ++q) {
        int r = r0 + q;
        float v;
        if (t < HDIM) {
            float d = deg[r]; d = (d < 1.f) ? 1.f : d;
            v = agg[(size_t)r * HDIM + t] / d;
        } else {
            v = h_in[(size_t)r * HDIM + (t - HDIM)];
        }
        Abf[q][t] = f2bf_us(v);
    }
    __syncthreads();

    const int w     = t >> 5;         // wave id = column tile
    const int lane  = t & 31;
    const int halfh = lane >> 4;      // 0: lanes 0-15, 1: lanes 16-31
    const int lm    = lane & 15;
    const int col   = w * 16 + lm;

    v8f acc;
    {
        float bias = bl[col];
        #pragma unroll
        for (int i = 0; i < 8; ++i) acc[i] = bias;
    }

    const unsigned short* brow = Wcat_t + col * 256;   // this lane's B column, K contiguous

    #pragma unroll
    for (int kt = 0; kt < 8; ++kt) {
        const int kbase = kt * 32;
        // A 16x32 bf16: lane(half,lm) holds elems i<8 -> K=kbase+8*half+i,
        //               i>=8 -> K=kbase+16+8*half+(i-8): two 16B contiguous runs.
        Q2 qa;
        qa.x = *(const uint4*)&Abf[lm][kbase + halfh * 8];
        qa.y = *(const uint4*)&Abf[lm][kbase + 16 + halfh * 8];
        v16bf afrag = __builtin_bit_cast(v16bf, qa);
        // B 32x16 bf16: same K order as A: two 16B runs at kbase+8*half and kbase+16+8*half.
        Q2 qb;
        qb.x = *(const uint4*)(brow + kbase + halfh * 8);
        qb.y = *(const uint4*)(brow + kbase + 16 + halfh * 8);
        v16bf bfrag = __builtin_bit_cast(v16bf, qb);

        acc = __builtin_amdgcn_wmma_f32_16x16x32_bf16(
                  false, afrag, false, bfrag, (short)0, acc, false, false);
    }

    // D 16x16 f32 layout: VGPR i, lane l -> row = i + 8*(l>>4), col = l&15
    #pragma unroll
    for (int i = 0; i < 8; ++i) Cls[i + 8 * halfh][col] = acc[i];
    __syncthreads();

    // row-wise LayerNorm stats
    if (t < 16) {
        float s = 0.f;
        for (int j = 0; j < HDIM; ++j) s += Cls[t][j];
        float m = s * (1.f / HDIM);
        float s2 = 0.f;
        for (int j = 0; j < HDIM; ++j) { float d = Cls[t][j] - m; s2 += d * d; }
        mu_s[t] = m;
        rs_s[t] = rsqrtf(s2 * (1.f / HDIM) + 1e-5f);
    }
    __syncthreads();

    // apply LN + relu + residual; thread handles 8 contiguous cols of one row
    const int row = t >> 4;
    const int c0  = (t & 15) * 8;
    const int r   = r0 + row;
    #pragma unroll
    for (int j = 0; j < 8; ++j) {
        int c = c0 + j;
        float v = (Cls[row][c] - mu_s[row]) * rs_s[row] * lng[c] + lnb[c];
        v = fmaxf(v, 0.f);
        h_out[(size_t)r * HDIM + c] = h_in[(size_t)r * HDIM + c] + v;
    }
}

// ---------------- per-graph node count (run-length, batch is sorted) ----------------
__global__ void k_cnt(const int* __restrict__ batch, float* __restrict__ cnt) {
    int chunk = blockIdx.x * blockDim.x + threadIdx.x;
    int n0 = chunk * 128;
    if (n0 >= N_NODES) return;
    int end = n0 + 128; if (end > N_NODES) end = N_NODES;
    int cur = batch[n0];
    float acc = 0.f;
    for (int n = n0; n < end; ++n) {
        int b = batch[n];
        if (b != cur) { atomicAdd(&cnt[cur], acc); cur = b; acc = 0.f; }
        acc += 1.f;
    }
    atomicAdd(&cnt[cur], acc);
}

// ------- segment mean-sum + max pooling: thread = (column, 32-node run), batch sorted -------
__global__ void k_pool(const float* __restrict__ h, const int* __restrict__ batch,
                       float* __restrict__ psum, unsigned* __restrict__ pmax) {
    int gid = blockIdx.x * blockDim.x + threadIdx.x;   // 128 cols * (N/32) chunks
    if (gid >= HDIM * (N_NODES / 32)) return;
    int c  = gid & 127;
    int n0 = (gid >> 7) * 32;
    int end = n0 + 32; if (end > N_NODES) end = N_NODES;
    int cur = batch[n0];
    float s = 0.f, m = -3.4e38f;
    for (int n = n0; n < end; ++n) {
        int b = batch[n];
        float v = h[(size_t)n * HDIM + c];
        if (b != cur) {
            atomicAdd(&psum[cur * HDIM + c], s);
            atomicMax(&pmax[cur * HDIM + c], f2o(m));
            cur = b; s = 0.f; m = -3.4e38f;
        }
        s += v; m = fmaxf(m, v);
    }
    atomicAdd(&psum[cur * HDIM + c], s);
    atomicMax(&pmax[cur * HDIM + c], f2o(m));
}

// ---------------- trackster encoder (1 block per graph, HT threads) ----------------
__global__ void k_ts(const float* __restrict__ ts, const float* __restrict__ W1,
                     const float* __restrict__ b1, const float* __restrict__ lng,
                     const float* __restrict__ lnb, const float* __restrict__ W2,
                     const float* __restrict__ b2, float* __restrict__ tout) {
    __shared__ float u[HTDIM], act[HTDIM];
    __shared__ float mu, rs;
    int b = blockIdx.x, j = threadIdx.x;
    float t0 = ts[b * 3 + 0], t1 = ts[b * 3 + 1], t2 = ts[b * 3 + 2];
    u[j] = b1[j] + t0 * W1[0 * HTDIM + j] + t1 * W1[1 * HTDIM + j] + t2 * W1[2 * HTDIM + j];
    __syncthreads();
    if (j == 0) {
        float s = 0.f;
        for (int k = 0; k < HTDIM; ++k) s += u[k];
        float m = s * (1.f / HTDIM);
        float s2 = 0.f;
        for (int k = 0; k < HTDIM; ++k) { float d = u[k] - m; s2 += d * d; }
        mu = m; rs = rsqrtf(s2 * (1.f / HTDIM) + 1e-5f);
    }
    __syncthreads();
    act[j] = fmaxf((u[j] - mu) * rs * lng[j] + lnb[j], 0.f);
    __syncthreads();
    float o = b2[j];
    for (int k = 0; k < HTDIM; ++k) o += act[k] * W2[k * HTDIM + j];
    tout[b * HTDIM + j] = o;
}

// ---------------- classifier head (1 block per graph, 128 threads) ----------------
__global__ __launch_bounds__(128) void k_cls(
    const float* __restrict__ psum, const unsigned* __restrict__ pmax,
    const float* __restrict__ cnt, const float* __restrict__ tin,
    const float* __restrict__ lng, const float* __restrict__ lnb,
    const float* __restrict__ W1, const float* __restrict__ b1,
    const float* __restrict__ W2, const float* __restrict__ b2,
    float* __restrict__ out) {
    __shared__ float f[PDIM], nf[PDIM], z[HDIM];
    __shared__ float mu, rs;
    int b = blockIdx.x, t = threadIdx.x;
    float c = cnt[b]; if (c < 1.f) c = 1.f;
    for (int i = t; i < PDIM; i += 128) {
        float v;
        if (i < HDIM)            v = psum[b * HDIM + i] / c;
        else if (i < 2 * HDIM)   v = o2f(pmax[b * HDIM + (i - HDIM)]);
        else                     v = tin[b * HTDIM + (i - 2 * HDIM)];
        f[i] = v;
    }
    __syncthreads();
    if (t == 0) {
        float s = 0.f;
        for (int k = 0; k < PDIM; ++k) s += f[k];
        float m = s * (1.f / PDIM);
        float s2 = 0.f;
        for (int k = 0; k < PDIM; ++k) { float d = f[k] - m; s2 += d * d; }
        mu = m; rs = rsqrtf(s2 * (1.f / PDIM) + 1e-5f);
    }
    __syncthreads();
    for (int i = t; i < PDIM; i += 128) nf[i] = (f[i] - mu) * rs * lng[i] + lnb[i];
    __syncthreads();
    float zz = b1[t];
    for (int k = 0; k < PDIM; ++k) zz += nf[k] * W1[k * HDIM + t];
    z[t] = fmaxf(zz, 0.f);
    __syncthreads();
    if (t < NCLS) {
        float o = b2[t];
        for (int k = 0; k < HDIM; ++k) o += z[k] * W2[k * NCLS + t];
        out[b * NCLS + t] = o;
    }
}

extern "C" void kernel_launch(void* const* d_in, const int* in_sizes, int n_in,
                              void* d_out, int out_size, void* d_ws, size_t ws_size,
                              hipStream_t stream) {
    const float* x      = (const float*)d_in[0];
    const int*   ei     = (const int*)  d_in[1];   // [2, E]: src then dst
    const int*   batch  = (const int*)  d_in[2];
    const float* ts     = (const float*)d_in[3];
    const float* encW   = (const float*)d_in[4];
    const float* encb   = (const float*)d_in[5];
    const float* sageWl = (const float*)d_in[6];
    const float* sagebl = (const float*)d_in[7];
    const float* sageWr = (const float*)d_in[8];
    const float* lng    = (const float*)d_in[9];
    const float* lnb    = (const float*)d_in[10];
    const float* tsW1   = (const float*)d_in[11];
    const float* tsb1   = (const float*)d_in[12];
    const float* tslng  = (const float*)d_in[13];
    const float* tslnb  = (const float*)d_in[14];
    const float* tsW2   = (const float*)d_in[15];
    const float* tsb2   = (const float*)d_in[16];
    const float* clslng = (const float*)d_in[17];
    const float* clslnb = (const float*)d_in[18];
    const float* clsW1  = (const float*)d_in[19];
    const float* clsb1  = (const float*)d_in[20];
    const float* clsW2  = (const float*)d_in[21];
    const float* clsb2  = (const float*)d_in[22];

    // workspace layout
    char* ws = (char*)d_ws;
    size_t off = 0;
    auto take = [&](size_t bytes) { size_t o = off; off = (off + bytes + 255) & ~(size_t)255; return o; };
    const size_t NH = (size_t)N_NODES * HDIM;
    float*          h0   = (float*)         (ws + take(NH * 4));
    float*          h1   = (float*)         (ws + take(NH * 4));
    float*          agg  = (float*)         (ws + take(NH * 4));
    float*          deg  = (float*)         (ws + take((size_t)N_NODES * 4));
    float*          cnt  = (float*)         (ws + take((size_t)BGRAPH * 4));
    float*          psum = (float*)         (ws + take((size_t)BGRAPH * HDIM * 4));
    unsigned*       pmax = (unsigned*)      (ws + take((size_t)BGRAPH * HDIM * 4));
    float*          tbuf = (float*)         (ws + take((size_t)BGRAPH * HTDIM * 4));
    unsigned short* Wcat = (unsigned short*)(ws + take((size_t)256 * HDIM * 2));

    const int* e_src = ei;
    const int* e_dst = ei + N_EDGES;

    // encoder + degree
    hipMemsetAsync(deg, 0, (size_t)N_NODES * 4, stream);
    k_encoder<<<(N_NODES * HDIM) / 256, 256, 0, stream>>>(x, encW, encb, h0);
    k_deg<<<(N_EDGES + 255) / 256, 256, 0, stream>>>(e_dst, deg);

    // SAGE layers (ping-pong h buffers)
    float* hc = h0;
    float* hn = h1;
    for (int i = 0; i < LAYERS; ++i) {
        hipMemsetAsync(agg, 0, NH * 4, stream);
        k_wcat<<<(256 * HDIM) / 256, 256, 0, stream>>>(
            sageWl + (size_t)i * HDIM * HDIM, sageWr + (size_t)i * HDIM * HDIM, Wcat);
        k_scatter<<<(N_EDGES * 16) / 256, 256, 0, stream>>>(e_src, e_dst, hc, agg);
        k_sage<<<N_NODES / 16, 256, 0, stream>>>(
            hc, agg, deg, Wcat, sagebl + i * HDIM, lng + i * HDIM, lnb + i * HDIM, hn);
        float* tmp = hc; hc = hn; hn = tmp;
    }
    // hc now holds the final node features

    // pooling
    hipMemsetAsync(psum, 0, (size_t)BGRAPH * HDIM * 4, stream);
    hipMemsetAsync(pmax, 0, (size_t)BGRAPH * HDIM * 4, stream);   // 0 < f2o(any real)
    hipMemsetAsync(cnt,  0, (size_t)BGRAPH * 4, stream);
    k_cnt<<<(N_NODES / 128 + 256) / 256 + 1, 256, 0, stream>>>(batch, cnt);
    {
        int pool_threads = HDIM * (N_NODES / 32);
        k_pool<<<(pool_threads + 255) / 256, 256, 0, stream>>>(hc, batch, psum, pmax);
    }

    // heads
    k_ts<<<BGRAPH, HTDIM, 0, stream>>>(ts, tsW1, tsb1, tslng, tslnb, tsW2, tsb2, tbuf);
    k_cls<<<BGRAPH, 128, 0, stream>>>(psum, pmax, cnt, tbuf, clslng, clslnb,
                                      clsW1, clsb1, clsW2, clsb2, (float*)d_out);
}
